// RingDilatedAttentionV3_64690797412682
// MI455X (gfx1250) — matrix operand order repaired and checked by reading, not compiled, not run on hardware
//
#include <hip/hip_runtime.h>
#include <hip/hip_bf16.h>

// MI455X / gfx1250 implementation. All matmuls (QKV proj, QK^T, P*V, out proj)
// go through v_wmma_f32_16x16x32_bf16 (wave32). Problem is bandwidth-bound
// (~116 GFLOP vs ~hundreds of MB of traffic at 23.3 TB/s), so we convert once
// to bf16 and keep everything resident in workspace (fits in 192MB L2).
// The attention kernel double-buffers its KV tiles and uses CDNA5
// GLOBAL_LOAD_ASYNC_TO_LDS_B128 (ASYNCcnt) to overlap DMA with WMMA.

typedef __attribute__((ext_vector_type(16))) __bf16 v16bf;
typedef __attribute__((ext_vector_type(8)))  __bf16 v8bf;
typedef __attribute__((ext_vector_type(8)))  float  v8f;

#define S_LEN 8192
#define EMB   768
#define NH    12
#define HD    64
#define N3    2304   // 3*EMB

static __device__ __forceinline__ __bf16 f2bf(float f) {
  unsigned u = __builtin_bit_cast(unsigned, f);
  u += 0x7fffu + ((u >> 16) & 1u);          // round-to-nearest-even
  unsigned short h = (unsigned short)(u >> 16);
  return __builtin_bit_cast(__bf16, h);
}

static __device__ __forceinline__ v16bf cat8(v8bf a, v8bf b) {
  return __builtin_shufflevector(a, b, 0,1,2,3,4,5,6,7,8,9,10,11,12,13,14,15);
}

static __device__ __forceinline__ v8f wmma_bf16(v16bf a, v16bf b, v8f c) {
  // 8 args: (neg_a, A, neg_b, B, c_mod, C, reuse_a, reuse_b)
  return __builtin_amdgcn_wmma_f32_16x16x32_bf16(false, a, false, b, (short)0, c,
                                                 false, false);
}

// 32-byte async copy global -> LDS, tracked by ASYNCcnt (2x b128 per lane).
static __device__ __forceinline__ void async_copy32(const __bf16* gsrc,
                                                    __bf16* ldst) {
  unsigned lds = (unsigned)(uintptr_t)ldst;           // LDS aperture: low 32b
  unsigned long long ga = (unsigned long long)(uintptr_t)gsrc;
  asm volatile("global_load_async_to_lds_b128 %0, %1, off\n\t"
               "global_load_async_to_lds_b128 %0, %1, off offset:16"
               :: "v"(lds), "v"(ga) : "memory");
}

static __device__ __forceinline__ void wait_async0() {
  asm volatile("s_wait_asynccnt 0x0" ::: "memory");
}

// ---------------------------------------------------------------- elementwise
__global__ void k_cvt_f32_bf16(const float* __restrict__ in,
                               __bf16* __restrict__ out, int n) {
  int i = blockIdx.x * blockDim.x + threadIdx.x;
  if (i < n) out[i] = f2bf(in[i]);
}

__global__ void k_zero_f32(float* __restrict__ p, int n) {
  int i = blockIdx.x * blockDim.x + threadIdx.x;
  if (i < n) p[i] = 0.0f;
}

// ---------------------------------------------------------------- QKV GEMM
// C[64x64] per 128-thread block. A: [S_LEN x EMB] bf16. B: [EMB x N3] bf16.
// Epilogue scatters (with bias) into Q/K/V laid out [H][S][D] bf16.
__global__ __launch_bounds__(128)
void qkv_gemm(const __bf16* __restrict__ A, const __bf16* __restrict__ B,
              const float* __restrict__ bias,
              __bf16* __restrict__ Qg, __bf16* __restrict__ Kg,
              __bf16* __restrict__ Vg) {
  __shared__ __align__(32) __bf16 As[64 * 32];   // [m][k]
  __shared__ __align__(32) __bf16 Bt[64 * 32];   // [n][k] (transposed)
  const int tid  = threadIdx.x;
  const int lane = tid & 31;
  const int w    = tid >> 5;
  const int hi   = lane >> 4;
  const int ln   = lane & 15;
  const int m0   = blockIdx.y * 64;
  const int n0   = blockIdx.x * 64;

  v8f acc[4] = {};
  for (int kt = 0; kt < EMB; kt += 32) {
    __syncthreads();
    { int row = tid >> 1, c0 = (tid & 1) * 16;
      *(v16bf*)(As + row * 32 + c0) =
          *(const v16bf*)(A + (size_t)(m0 + row) * EMB + kt + c0); }
    { int k = tid >> 2, c0 = (tid & 3) * 16;
      v16bf v = *(const v16bf*)(B + (size_t)(kt + k) * N3 + n0 + c0);
      #pragma unroll
      for (int e = 0; e < 16; ++e) Bt[(c0 + e) * 32 + k] = v[e]; }
    __syncthreads();
    // A fragment: 16x32, rows 16w..16w+15 (lane m = ln), k = 16*hi + s (contiguous, same order as B)
    v16bf a = *(const v16bf*)(As + (16 * w + ln) * 32 + 16 * hi);
    #pragma unroll
    for (int t = 0; t < 4; ++t) {
      v16bf b = *(const v16bf*)(Bt + (16 * t + ln) * 32 + 16 * hi);
      acc[t] = wmma_bf16(a, b, acc[t]);
    }
  }
  #pragma unroll
  for (int t = 0; t < 4; ++t) {
    int n     = n0 + 16 * t + ln;
    int three = n / EMB;
    int h     = (n % EMB) / HD;
    int d     = n % HD;
    __bf16* dst = (three == 0) ? Qg : (three == 1) ? Kg : Vg;
    float bv = bias[n];
    #pragma unroll
    for (int r = 0; r < 8; ++r) {
      int m = m0 + 16 * w + r + 8 * hi;
      dst[((size_t)h * S_LEN + m) * HD + d] = f2bf(acc[t][r] + bv);
    }
  }
}

// ---------------------------------------------------------------- attention
// Flash attention over the dilated index set {dil*i : i < L}. One block =
// 4 waves = 64 query rows of one head; KV streamed in 32-wide tiles.
// Double-buffered: tile it+1's K is DMA'd via async-to-LDS while tile it is
// consumed by WMMA; V for tile it+1 rides in VGPRs and is transposed into
// LDS after the current tile's P*V.
__global__ __launch_bounds__(128)
void dilated_attn(const __bf16* __restrict__ Qg, const __bf16* __restrict__ Kg,
                  const __bf16* __restrict__ Vg, float* __restrict__ accum,
                  int L, int dil, float wgt) {
  __shared__ __align__(32) __bf16 Kl[2][32 * 64];     // [kv][d]
  __shared__ __align__(32) __bf16 Vt[2][64 * 32];     // [d][kv] (transposed)
  __shared__ __align__(32) __bf16 Pl[4][16 * 32];     // per-wave P staging
  const int tid  = threadIdx.x;
  const int lane = tid & 31;
  const int w    = tid >> 5;
  const int hi   = lane >> 4;
  const int ln   = lane & 15;
  const int h    = blockIdx.y;
  const int q0   = blockIdx.x * 64;
  const size_t headOff = (size_t)h * S_LEN * HD;
  const int j  = tid >> 2;          // kv row this thread copies
  const int c0 = (tid & 3) * 16;    // d-chunk this thread copies

  // Q fragments (16 rows x 64 d) for this wave, held in registers
  const int qrow = q0 + 16 * w + ln;
  const __bf16* qp = Qg + headOff + (size_t)(dil * qrow) * HD;
  v16bf qa0 = *(const v16bf*)(qp + 16 * hi);
  v16bf qa1 = *(const v16bf*)(qp + 32 + 16 * hi);

  float mstate[8], lstate[8];
  v8f o[4] = {};
  #pragma unroll
  for (int r = 0; r < 8; ++r) { mstate[r] = -3.0e38f; lstate[r] = 0.0f; }

  // ---- preload tile 0 into buffer 0
  {
    size_t rowOff = headOff + (size_t)(dil * j) * HD + c0;
    async_copy32(Kg + rowOff, &Kl[0][j * 64 + c0]);
    v16bf v0 = *(const v16bf*)(Vg + rowOff);
    #pragma unroll
    for (int e = 0; e < 16; ++e) Vt[0][(c0 + e) * 32 + j] = v0[e];
    wait_async0();
  }
  __syncthreads();

  const int nt = L / 32;
  int p = 0;
  for (int it = 0; it < nt; ++it) {
    const bool pre = (it + 1 < nt);
    v16bf vv = {};
    if (pre) {  // kick off next tile: K via async DMA, V into VGPRs
      size_t rowOff = headOff + (size_t)(dil * ((it + 1) * 32 + j)) * HD + c0;
      async_copy32(Kg + rowOff, &Kl[1 - p][j * 64 + c0]);
      vv = *(const v16bf*)(Vg + rowOff);
    }
    const __bf16* Kp = Kl[p];
    const __bf16* Vp = Vt[p];

    // scores: S = Q K^T, two 16x16 tiles (kv cols 0-15 and 16-31)
    v8f s0 = {}, s1 = {};
    { v16bf b0 = *(const v16bf*)(Kp + ln * 64 + 16 * hi);
      v16bf b1 = *(const v16bf*)(Kp + ln * 64 + 32 + 16 * hi);
      s0 = wmma_bf16(qa0, b0, s0);
      s0 = wmma_bf16(qa1, b1, s0);
      v16bf b2 = *(const v16bf*)(Kp + (16 + ln) * 64 + 16 * hi);
      v16bf b3 = *(const v16bf*)(Kp + (16 + ln) * 64 + 32 + 16 * hi);
      s1 = wmma_bf16(qa0, b2, s1);
      s1 = wmma_bf16(qa1, b3, s1); }

    // online softmax (row r+8*hi lives in vgpr r across the 16-lane half)
    float p0[8], p1[8];
    #pragma unroll
    for (int r = 0; r < 8; ++r) {
      float a0 = s0[r] * 0.125f;   // 1/sqrt(64)
      float a1 = s1[r] * 0.125f;
      float v = fmaxf(a0, a1);
      v = fmaxf(v, __shfl_xor(v, 1));
      v = fmaxf(v, __shfl_xor(v, 2));
      v = fmaxf(v, __shfl_xor(v, 4));
      v = fmaxf(v, __shfl_xor(v, 8));
      float nm = fmaxf(mstate[r], v);
      float sc = __expf(mstate[r] - nm);
      mstate[r] = nm;
      p0[r] = __expf(a0 - nm);
      p1[r] = __expf(a1 - nm);
      float rs = p0[r] + p1[r];
      rs += __shfl_xor(rs, 1);
      rs += __shfl_xor(rs, 2);
      rs += __shfl_xor(rs, 4);
      rs += __shfl_xor(rs, 8);
      lstate[r] = lstate[r] * sc + rs;
      #pragma unroll
      for (int t = 0; t < 4; ++t) o[t][r] *= sc;
    }

    // restage P (D-layout f32 -> A-layout bf16) through per-wave LDS
    __bf16* pw = Pl[w];
    #pragma unroll
    for (int r = 0; r < 8; ++r) {
      pw[(r + 8 * hi) * 32 + ln]      = f2bf(p0[r]);
      pw[(r + 8 * hi) * 32 + 16 + ln] = f2bf(p1[r]);
    }
    __syncthreads();
    v16bf pa = *(const v16bf*)(pw + ln * 32 + 16 * hi);
    #pragma unroll
    for (int t = 0; t < 4; ++t) {
      v16bf vb = *(const v16bf*)(Vp + (16 * t + ln) * 32 + 16 * hi);
      o[t] = wmma_bf16(pa, vb, o[t]);
    }

    if (pre) {  // transpose next tile's V into the spare buffer
      #pragma unroll
      for (int e = 0; e < 16; ++e) Vt[1 - p][(c0 + e) * 32 + j] = vv[e];
    }
    wait_async0();    // next K tile is in LDS
    __syncthreads();  // ... and visible to all waves; all readers of p done
    p ^= 1;
  }

  // epilogue: scatter wgt * O / l into f32 accumulator at dilated positions
  #pragma unroll
  for (int r = 0; r < 8; ++r) {
    float inv = wgt / lstate[r];
    int m = q0 + 16 * w + r + 8 * hi;
    size_t pos = (size_t)dil * m;
    #pragma unroll
    for (int t = 0; t < 4; ++t) {
      float* ap = accum + pos * EMB + h * HD + 16 * t + ln;
      *ap += o[t][r] * inv;
    }
  }
}

// ---------------------------------------------------------------- out GEMM
__global__ __launch_bounds__(128)
void out_gemm(const __bf16* __restrict__ A, const __bf16* __restrict__ B,
              const float* __restrict__ bias, float* __restrict__ out) {
  __shared__ __align__(32) __bf16 As[64 * 32];
  __shared__ __align__(32) __bf16 Bt[64 * 32];
  const int tid  = threadIdx.x;
  const int lane = tid & 31;
  const int w    = tid >> 5;
  const int hi   = lane >> 4;
  const int ln   = lane & 15;
  const int m0   = blockIdx.y * 64;
  const int n0   = blockIdx.x * 64;

  v8f acc[4] = {};
  for (int kt = 0; kt < EMB; kt += 32) {
    __syncthreads();
    { int row = tid >> 1, c0 = (tid & 1) * 16;
      *(v16bf*)(As + row * 32 + c0) =
          *(const v16bf*)(A + (size_t)(m0 + row) * EMB + kt + c0); }
    { int k = tid >> 2, c0 = (tid & 3) * 16;
      v16bf v = *(const v16bf*)(B + (size_t)(kt + k) * EMB + n0 + c0);
      #pragma unroll
      for (int e = 0; e < 16; ++e) Bt[(c0 + e) * 32 + k] = v[e]; }
    __syncthreads();
    v16bf a = *(const v16bf*)(As + (16 * w + ln) * 32 + 16 * hi);
    #pragma unroll
    for (int t = 0; t < 4; ++t) {
      v16bf b = *(const v16bf*)(Bt + (16 * t + ln) * 32 + 16 * hi);
      acc[t] = wmma_bf16(a, b, acc[t]);
    }
  }
  #pragma unroll
  for (int t = 0; t < 4; ++t) {
    int n = n0 + 16 * t + ln;
    float bv = bias[n];
    #pragma unroll
    for (int r = 0; r < 8; ++r) {
      int m = m0 + 16 * w + r + 8 * hi;
      out[(size_t)m * EMB + n] = acc[t][r] + bv;
    }
  }
}

// ---------------------------------------------------------------- launch
extern "C" void kernel_launch(void* const* d_in, const int* in_sizes, int n_in,
                              void* d_out, int out_size, void* d_ws,
                              size_t ws_size, hipStream_t stream) {
  (void)in_sizes; (void)n_in; (void)out_size; (void)ws_size;
  const float* x    = (const float*)d_in[0];
  const float* Wqkv = (const float*)d_in[1];
  const float* bqkv = (const float*)d_in[2];
  const float* Wout = (const float*)d_in[3];
  const float* bout = (const float*)d_in[4];
  float* out = (float*)d_out;

  char* ws = (char*)d_ws;
  const size_t SZ_X    = (size_t)S_LEN * EMB * 2;   // 12,582,912
  const size_t SZ_WQKV = (size_t)EMB * N3 * 2;      //  3,538,944
  const size_t SZ_WOUT = (size_t)EMB * EMB * 2;     //  1,179,648
  const size_t SZ_HSD  = (size_t)NH * S_LEN * HD * 2;
  size_t off = 0;
  __bf16* xb    = (__bf16*)(ws + off); off += SZ_X;      // also reused for attn
  __bf16* wqkvb = (__bf16*)(ws + off); off += SZ_WQKV;
  __bf16* woutb = (__bf16*)(ws + off); off += SZ_WOUT;
  __bf16* Qg    = (__bf16*)(ws + off); off += SZ_HSD;
  __bf16* Kg    = (__bf16*)(ws + off); off += SZ_HSD;
  __bf16* Vg    = (__bf16*)(ws + off); off += SZ_HSD;
  float*  acc   = (float*)(ws + off);  off += (size_t)S_LEN * EMB * 4;
  __bf16* attnb = xb;  // x is dead after qkv_gemm; alias its region

  const int NX = S_LEN * EMB;
  k_cvt_f32_bf16<<<(NX + 255) / 256, 256, 0, stream>>>(x, xb, NX);
  k_cvt_f32_bf16<<<(EMB * N3 + 255) / 256, 256, 0, stream>>>(Wqkv, wqkvb, EMB * N3);
  k_cvt_f32_bf16<<<(EMB * EMB + 255) / 256, 256, 0, stream>>>(Wout, woutb, EMB * EMB);

  qkv_gemm<<<dim3(N3 / 64, S_LEN / 64), 128, 0, stream>>>(xb, wqkvb, bqkv,
                                                          Qg, Kg, Vg);

  k_zero_f32<<<(NX + 255) / 256, 256, 0, stream>>>(acc, NX);

  const float W3 = 1.0f / 3.0f;
  dilated_attn<<<dim3(2048 / 64, NH), 128, 0, stream>>>(Qg, Kg, Vg, acc, 2048, 1, W3);
  dilated_attn<<<dim3(4096 / 64, NH), 128, 0, stream>>>(Qg, Kg, Vg, acc, 4096, 2, W3);
  dilated_attn<<<dim3(2048 / 64, NH), 128, 0, stream>>>(Qg, Kg, Vg, acc, 2048, 4, W3);

  k_cvt_f32_bf16<<<(NX + 255) / 256, 256, 0, stream>>>(acc, attnb, NX);

  out_gemm<<<dim3(EMB / 64, S_LEN / 64), 128, 0, stream>>>(attnb, woutb, bout, out);
}
